// MultiHeadGraphAttentionModule_47347719471736
// MI455X (gfx1250) — compile-verified, matrix operand order checked by reading, not run on hardware
//
#include <hip/hip_runtime.h>

// MI455X / gfx1250 fused multi-head GAT.
//   in[0] features  [4096, 512]  f32
//   in[1] adjacency [4096, 4096] i32
//   in[2] W         [8, 512, 64] f32
//   in[3] a_src     [8, 64]      f32
//   in[4] a_dst     [8, 64]      f32
//   out             [4096, 512]  f32   ( [N, H*F_out] )
//
// Strategy: bf16 WMMA (f32 accumulate) for both GEMMs; flash-attention-style
// online softmax so the 4.3 GB attention tensor is never materialized. The
// Tensor Data Mover DMAs the (transposed) Wh j-tile into LDS with hardware
// row padding (144 B stride -> bank-conflict-free b128 fragment loads),
// issued before the softmax VALU work and waited with s_wait_tensorcnt just
// before the WMMAs, so the DMA hides under the exp/mask work that dominates
// the roofline (~15us VALU vs ~10us HBM; adjacency is L2-resident after
// head 0 touches it: 64 MB < 192 MB L2).

typedef __attribute__((ext_vector_type(16))) __bf16 v16bf;
typedef __attribute__((ext_vector_type(8)))  float  v8f;
typedef __attribute__((ext_vector_type(4)))  unsigned int v4u;
typedef __attribute__((ext_vector_type(8)))  int    v8i;
typedef __attribute__((ext_vector_type(4)))  int    v4i;

#define N_NODES 4096
#define F_IN    512
#define F_OUT   64
#define HEADS   8
#define NEG_INF (-1e9f)
#define ALPHA   0.2f

// Padded LDS row strides (in bf16 elements): 144 B = 36 dwords; lane*36 mod 64
// covers 16 distinct 4-bank groups -> conflict-free ds_load_b128 fragments.
#define BT_STRIDE 72
#define P_STRIDE  72

#if defined(__has_builtin)
#  if __has_builtin(__builtin_amdgcn_tensor_load_to_lds) && \
      __has_builtin(__builtin_amdgcn_s_wait_tensorcnt)
#    define HAVE_TDM 1
#  endif
#endif
#ifndef HAVE_TDM
#  define HAVE_TDM 0
#endif

// ---------------------------------------------------------------------------
// Kernel 1: per-head projection  Wh[h] = features @ W[h]   (bf16 WMMA, f32 acc)
// grid = HEADS * (N/16) blocks of 128 threads (4 waves); wave w owns the
// 16x16 output tile at columns [16w, 16w+16). Stores f32 row-major (for the
// src/dst dot products) and bf16 TRANSPOSED [H][F_OUT][N] (so the attention
// B-operand K-runs are contiguous).
// ---------------------------------------------------------------------------
__global__ __launch_bounds__(128) void gat_proj_kernel(
    const float* __restrict__ features,   // [N, F_IN]
    const float* __restrict__ W,          // [H, F_IN, F_OUT]
    float*       __restrict__ Wh,         // [H, N, F_OUT] f32
    __bf16*      __restrict__ WhBT)       // [H, F_OUT, N] bf16 (transposed)
{
  const int blk     = blockIdx.x;
  const int h       = blk / (N_NODES / 16);
  const int rowBase = (blk % (N_NODES / 16)) * 16;
  const int wave    = threadIdx.x >> 5;
  const int lane    = threadIdx.x & 31;
  const int half    = lane >> 4;          // K-half select per ISA A/B layout
  const int mrow    = lane & 15;          // M for A-frag, N for B-frag
  const int colBase = wave * 16;

  v8f c = {};
  for (int k0 = 0; k0 < F_IN; k0 += 32) {
    // A fragment: features[rowBase+mrow][k0 + half*8 + {0..7}] and +16
    const float* arow = features + (size_t)(rowBase + mrow) * F_IN + k0 + half * 8;
    const float4 a0 = *(const float4*)(arow);
    const float4 a1 = *(const float4*)(arow + 4);
    const float4 a2 = *(const float4*)(arow + 16);
    const float4 a3 = *(const float4*)(arow + 20);
    v16bf a;
    a[0]  = (__bf16)a0.x; a[1]  = (__bf16)a0.y; a[2]  = (__bf16)a0.z; a[3]  = (__bf16)a0.w;
    a[4]  = (__bf16)a1.x; a[5]  = (__bf16)a1.y; a[6]  = (__bf16)a1.z; a[7]  = (__bf16)a1.w;
    a[8]  = (__bf16)a2.x; a[9]  = (__bf16)a2.y; a[10] = (__bf16)a2.z; a[11] = (__bf16)a2.w;
    a[12] = (__bf16)a3.x; a[13] = (__bf16)a3.y; a[14] = (__bf16)a3.z; a[15] = (__bf16)a3.w;

    // B fragment: W[h][k0 + half*8 + {0..7}][colBase+mrow] and K+16
    const float* wcol = W + ((size_t)h * F_IN + k0 + half * 8) * F_OUT + colBase + mrow;
    v16bf b;
#pragma unroll
    for (int i = 0; i < 8; ++i) {
      b[i]     = (__bf16)wcol[(size_t)i * F_OUT];
      b[i + 8] = (__bf16)wcol[(size_t)(16 + i) * F_OUT];
    }
    c = __builtin_amdgcn_wmma_f32_16x16x32_bf16(false, a, false, b,
                                                (short)0, c, false, false);
  }

  // C/D layout: VGPR r -> row r + 8*half, column = lane&15
#pragma unroll
  for (int r = 0; r < 8; ++r) {
    const int row = rowBase + r + 8 * half;
    Wh  [((size_t)h * N_NODES + row) * F_OUT + colBase + mrow] = c[r];
    WhBT[((size_t)h * F_OUT + colBase + mrow) * N_NODES + row] = (__bf16)c[r];
  }
}

// ---------------------------------------------------------------------------
// Kernel 2: src[h,n] = Wh[h,n,:].a_src[h]  ;  dst likewise   (8 MB read, tiny)
// ---------------------------------------------------------------------------
__global__ __launch_bounds__(256) void gat_vec_kernel(
    const float* __restrict__ Wh,
    const float* __restrict__ a_src, const float* __restrict__ a_dst,
    float* __restrict__ srcp, float* __restrict__ dstp)
{
  const int idx = blockIdx.x * blockDim.x + threadIdx.x;   // h*N + n
  if (idx >= HEADS * N_NODES) return;
  const int h = idx / N_NODES;
  const float* row = Wh + (size_t)idx * F_OUT;
  const float* as  = a_src + h * F_OUT;
  const float* ad  = a_dst + h * F_OUT;
  float s = 0.f, d = 0.f;
#pragma unroll 8
  for (int o = 0; o < F_OUT; ++o) { const float v = row[o]; s += v * as[o]; d += v * ad[o]; }
  srcp[idx] = s;
  dstp[idx] = d;
}

// ---------------------------------------------------------------------------
// Kernel 3: fused masked-softmax attention + attn@Wh + ELU (flash style).
// grid = HEADS * (N/16) blocks of 128 threads; block owns 16 query rows of
// one head; 4 waves each own 16 output columns. Per 64-column chunk:
//   wave0 issues a TDM DMA of WhBT[h][:, j0:j0+64] -> padded LDS tile
//   all threads: logits + mask + online-softmax stats + bf16 P tile in LDS
//   wave0 s_wait_tensorcnt, barrier, 2x v_wmma_f32_16x16x32_bf16 per wave
//   with all operands as conflict-free ds_load_b128 fragments.
// ---------------------------------------------------------------------------
__global__ __launch_bounds__(128) void gat_attn_kernel(
    const int*    __restrict__ adj,    // [N, N]
    const __bf16* __restrict__ WhBT,   // [H, F_OUT, N]
    const float*  __restrict__ srcp,   // [H, N]
    const float*  __restrict__ dstp,   // [H, N]
    float*        __restrict__ out)    // [N, H*F_OUT]
{
  __shared__ float  sm_src[16];
  __shared__ float  sm_m[16];
  __shared__ float  sm_l[16];
  __shared__ float  sm_scale[16];
  __shared__ float  sm_red[16][8];
  __shared__ __align__(16) __bf16 sm_P [16 * P_STRIDE];   // P tile  (2.25 KB)
  __shared__ __align__(16) __bf16 sm_BT[64 * BT_STRIDE];  // Wh tile (9 KB)

  const int blk     = blockIdx.x;
  const int h       = blk / (N_NODES / 16);
  const int rowBase = (blk % (N_NODES / 16)) * 16;
  const int wave    = threadIdx.x >> 5;
  const int lane    = threadIdx.x & 31;
  const int half    = lane >> 4;
  const int mrow    = lane & 15;
  const int colBase = wave * 16;

  const int row = threadIdx.x >> 3;    // e-tile mapping: 16 rows x 8 threads
  const int sub = threadIdx.x & 7;     // 8 contiguous columns each

  if (threadIdx.x < 16) {
    sm_m[threadIdx.x]   = NEG_INF;
    sm_l[threadIdx.x]   = 0.f;
    sm_src[threadIdx.x] = srcp[(size_t)h * N_NODES + rowBase + threadIdx.x];
  }
  __syncthreads();

  v8f c = {};
  const int* adjrow = adj + (size_t)(rowBase + row) * N_NODES;

  for (int j0 = 0; j0 < N_NODES; j0 += 64) {
#if HAVE_TDM
    // ---- TDM: DMA WhBT[h][0:64][j0:j0+64] into sm_BT with HW padding ----
    // tile_dim0=64 (contiguous K run), tile_dim1=64 rows, row stride 4096;
    // pad_enable, pad_interval=4 (every 32 dwords), pad_amount=3 (4 dwords)
    // -> LDS rows land at 144 B stride (BT_STRIDE).
    if (wave == 0) {
      const unsigned long long ga =
          (unsigned long long)(const void*)(WhBT + (size_t)h * F_OUT * N_NODES + j0);
      // LDS aperture keeps the byte offset in addr[31:0] (ISA 10.2)
      const unsigned lds_off = (unsigned)(size_t)(void*)sm_BT;
      v4u g0;
      g0.x = 1u;                                     // count=1, user-mode D#
      g0.y = lds_off;                                // lds_addr
      g0.z = (unsigned)(ga & 0xFFFFFFFFu);           // global_addr[31:0]
      g0.w = (unsigned)((ga >> 32) & 0x01FFFFFFu)    // global_addr[56:32]
           | (2u << 30);                             // type=2 ("image")
      v8i g1;
      g1[0] = (int)((1u << 16)                       // data_size=1 (2 bytes)
                  | (1u << 20)                       // pad_enable
                  | (4u << 22)                       // pad_interval: 32 dwords
                  | (3u << 25));                     // pad_amount: 4 dwords
      g1[1] = (int)(4096u << 16);                    // tensor_dim0 = 4096
      g1[2] = (int)(64u << 16);                      // tensor_dim1 = 64
      g1[3] = (int)(64u << 16);                      // tile_dim0 = 64
      g1[4] = 64;                                    // tile_dim1 = 64, tile_dim2=0
      g1[5] = 4096;                                  // tensor_dim0_stride = 4096
      g1[6] = 0;
      g1[7] = 0;
      const v4i gz = {};
#if __clang_major__ >= 23
      const v8i gz8 = {};
      __builtin_amdgcn_tensor_load_to_lds(g0, g1, gz, gz, gz8, 0);
#else
      __builtin_amdgcn_tensor_load_to_lds(g0, g1, gz, gz, 0);
#endif
    }
#else
    // Fallback: cooperative coalesced copy into the same padded layout
    {
      const __bf16* gsrc = WhBT + (size_t)h * F_OUT * N_NODES + j0;
      for (int i = threadIdx.x; i < 64 * 8; i += 128) {
        const int r = i >> 3, q = i & 7;
        const uint4 v = *(const uint4*)(gsrc + (size_t)r * N_NODES + q * 8);
        *(uint4*)((char*)sm_BT + r * (BT_STRIDE * 2) + q * 16) = v;
      }
    }
#endif

    // ---- logits for this 16x64 tile ------------------------------------
    const int jj = j0 + sub * 8;
    const int4   av0 = *(const int4*)  (adjrow + jj);
    const int4   av1 = *(const int4*)  (adjrow + jj + 4);
    const float4 dv0 = *(const float4*)(dstp + (size_t)h * N_NODES + jj);
    const float4 dv1 = *(const float4*)(dstp + (size_t)h * N_NODES + jj + 4);
    __builtin_prefetch(adjrow + jj + 64, 0, 0);           // global_prefetch_b8

    const float si = sm_src[row];
    int   am[8] = {av0.x, av0.y, av0.z, av0.w, av1.x, av1.y, av1.z, av1.w};
    float dj[8] = {dv0.x, dv0.y, dv0.z, dv0.w, dv1.x, dv1.y, dv1.z, dv1.w};
    float e[8];
#pragma unroll
    for (int i = 0; i < 8; ++i) {
      const float x = si + dj[i];
      e[i] = (am[i] != 0) ? (x > 0.f ? x : ALPHA * x) : NEG_INF;
    }
    float lm = e[0];
#pragma unroll
    for (int i = 1; i < 8; ++i) lm = fmaxf(lm, e[i]);
    sm_red[row][sub] = lm;
    __syncthreads();                                      // (A)

    // ---- per-row max + rescale factor ----------------------------------
    if (threadIdx.x < 16) {
      float cm = sm_red[threadIdx.x][0];
#pragma unroll
      for (int i = 1; i < 8; ++i) cm = fmaxf(cm, sm_red[threadIdx.x][i]);
      const float mo = sm_m[threadIdx.x];
      const float mn = fmaxf(mo, cm);
      sm_m[threadIdx.x]     = mn;
      sm_scale[threadIdx.x] = __expf(mo - mn);
    }
    __syncthreads();                                      // (B)

    // ---- probabilities (bf16 into LDS) + partial row sums --------------
    const float mn = sm_m[row];
    float psum = 0.f;
    __bf16* pp = sm_P + row * P_STRIDE + sub * 8;
#pragma unroll
    for (int i = 0; i < 8; ++i) {
      const float p = (am[i] != 0) ? __expf(e[i] - mn) : 0.f;
      pp[i] = (__bf16)p;
      psum += p;
    }
    sm_red[row][sub] = psum;

    // rescale running accumulator (exact online softmax)
#pragma unroll
    for (int r = 0; r < 8; ++r) c[r] *= sm_scale[r + 8 * half];

#if HAVE_TDM
    if (wave == 0) __builtin_amdgcn_s_wait_tensorcnt(0);  // sm_BT in LDS
#endif
    __syncthreads();                                      // (C)

    if (threadIdx.x < 16) {
      float s = sm_red[threadIdx.x][0];
#pragma unroll
      for (int i = 1; i < 8; ++i) s += sm_red[threadIdx.x][i];
      sm_l[threadIdx.x] = sm_l[threadIdx.x] * sm_scale[threadIdx.x] + s;
    }

    // ---- WMMA: acc += P(16x64) @ WhT-tile; b128 LDS fragments ----------
#pragma unroll
    for (int kk = 0; kk < 64; kk += 32) {
      v16bf a;
      const __bf16* prow = sm_P + mrow * P_STRIDE + kk + half * 8;
#pragma unroll
      for (int i = 0; i < 8; ++i) { a[i] = prow[i]; a[i + 8] = prow[16 + i]; }
      v16bf b;   // K-run is contiguous in the transposed tile
      const __bf16* brow = sm_BT + (colBase + mrow) * BT_STRIDE + kk + half * 8;
#pragma unroll
      for (int i = 0; i < 8; ++i) { b[i] = brow[i]; b[i + 8] = brow[16 + i]; }
      c = __builtin_amdgcn_wmma_f32_16x16x32_bf16(false, a, false, b,
                                                  (short)0, c, false, false);
    }
    __syncthreads();                                      // (D)
  }

  // ---- epilogue: normalize, ELU, scatter to [N, H*F_OUT] ---------------
#pragma unroll
  for (int r = 0; r < 8; ++r) {
    const int rr = r + 8 * half;
    float v = c[r] / sm_l[rr];
    v = (v > 0.f) ? v : (__expf(v) - 1.f);
    out[(size_t)(rowBase + rr) * (HEADS * F_OUT) + h * F_OUT + colBase + mrow] = v;
  }
}

// ---------------------------------------------------------------------------
extern "C" void kernel_launch(void* const* d_in, const int* in_sizes, int n_in,
                              void* d_out, int out_size, void* d_ws, size_t ws_size,
                              hipStream_t stream) {
  const float* features = (const float*)d_in[0];
  const int*   adj      = (const int*)  d_in[1];
  const float* W        = (const float*)d_in[2];
  const float* a_src    = (const float*)d_in[3];
  const float* a_dst    = (const float*)d_in[4];
  float* out = (float*)d_out;

  // Workspace carve-up (~12.25 MB total)
  char* ws = (char*)d_ws;
  float*  Wh   = (float*) (ws);                                         // 8 MB
  __bf16* WhBT = (__bf16*)(ws + (size_t)HEADS * N_NODES * F_OUT * 4);   // 4 MB
  float*  srcp = (float*) (ws + (size_t)HEADS * N_NODES * F_OUT * 6);   // 128 KB
  float*  dstp = srcp + (size_t)HEADS * N_NODES;                        // 128 KB

  const int nBlk = HEADS * (N_NODES / 16);   // 2048

  gat_proj_kernel<<<nBlk, 128, 0, stream>>>(features, W, Wh, WhBT);
  gat_vec_kernel<<<(HEADS * N_NODES) / 256, 256, 0, stream>>>(Wh, a_src, a_dst, srcp, dstp);
  gat_attn_kernel<<<nBlk, 128, 0, stream>>>(adj, WhBT, srcp, dstp, out);
}
